// TransformSingle_70626442215467
// MI455X (gfx1250) — compile-verified, matrix operand order checked by reading, not run on hardware
//
#include <hip/hip_runtime.h>
#include <hip/hip_bf16.h>

#define B_    2
#define S_    2048
#define D_    512
#define H_    8
#define DH_   64
#define ROWS_ (B_ * S_)   // 4096

typedef __attribute__((ext_vector_type(16))) _Float16 v16h;
typedef __attribute__((ext_vector_type(8)))  float    v8f;
typedef __attribute__((ext_vector_type(4)))  int      v4i;

#define WMMA_F16(a, b, c) \
  __builtin_amdgcn_wmma_f32_16x16x32_f16(false, (a), false, (b), (short)0, (c), false, false)

union FragA { v16h v; uint4 q[2]; _Float16 h[16]; };
union FragC { v8f  v; float f[8]; };

__device__ __forceinline__ int lane_id() { return (int)(threadIdx.x & 31u); }

// ---- CDNA5 async global->LDS copy (ASYNCcnt path), guarded so compile never breaks.
__device__ __forceinline__ void async_copy16(const _Float16* g, _Float16* l) {
#if __has_builtin(__builtin_amdgcn_global_load_async_to_lds_b128)
  __builtin_amdgcn_global_load_async_to_lds_b128(
      (v4i*)(void*)g,
      (__attribute__((address_space(3))) v4i*)(void*)l, 0, 0);
#else
  *reinterpret_cast<uint4*>(l) = *reinterpret_cast<const uint4*>(g);
#endif
}

__device__ __forceinline__ void async_wait0() {
#if __has_builtin(__builtin_amdgcn_s_wait_asynccnt)
  __builtin_amdgcn_s_wait_asynccnt(0);
#elif defined(__gfx1250__)
  asm volatile("s_wait_asynccnt 0x0" ::: "memory");
#endif
}

// Row-major 16-bit fragment load (A side: lane = M row; B side: lane = N col,
// source stored [N, K] row-major). Matches CDNA5 16-bit A layout:
// lane L holds row L&15, halves 0..7 = K c0..c0+7, halves 8..15 = K c0+16..c0+23,
// c0 = (L<16 ? 0 : 8). Works for both global and LDS source pointers.
__device__ __forceinline__ v16h frag_ld(const _Float16* __restrict__ p, int ld) {
  int lane = lane_id();
  const _Float16* q = p + (size_t)(lane & 15) * ld + ((lane & 16) ? 8 : 0);
  FragA u;
  u.q[0] = *reinterpret_cast<const uint4*>(q);
  u.q[1] = *reinterpret_cast<const uint4*>(q + 16);
  return u.v;
}

// Same fragment pattern sourced from the f32 LDS strip (ld = S_), cvt->f16.
__device__ __forceinline__ v16h frag_ld_lds_f32(const float* __restrict__ strip, int kbase) {
  int lane = lane_id();
  const float* p = strip + (size_t)(lane & 15) * S_ + kbase + ((lane & 16) ? 8 : 0);
  FragA u;
#pragma unroll
  for (int i = 0; i < 8; ++i) {
    u.h[i]     = (_Float16)p[i];
    u.h[i + 8] = (_Float16)p[16 + i];
  }
  return u.v;
}

__device__ __forceinline__ void frag_zero(FragC& c) {
#pragma unroll
  for (int i = 0; i < 8; ++i) c.f[i] = 0.0f;
}

// ---------------- K0: weight convert f32 [Din,Dout] -> f16 transposed [Dout,Din]
__global__ __launch_bounds__(256) void wconv(const float* __restrict__ src,
                                             _Float16* __restrict__ dst,
                                             int din, int dout) {
  int idx = blockIdx.x * blockDim.x + threadIdx.x;
  if (idx >= din * dout) return;
  int r = idx / dout, c = idx - r * dout;
  dst[(size_t)c * din + r] = (_Float16)src[idx];
}

// ---------------- K1/K5: layernorm over D=512, f32 in -> f16 out. 128 threads/row.
__global__ __launch_bounds__(128) void ln512(const float* __restrict__ x,
                                             const float* __restrict__ w,
                                             const float* __restrict__ bia,
                                             _Float16* __restrict__ out) {
  int row = blockIdx.x;
  int t = threadIdx.x;
  const float4 xv = reinterpret_cast<const float4*>(x + (size_t)row * D_)[t];
  float s  = xv.x + xv.y + xv.z + xv.w;
  float ss = xv.x * xv.x + xv.y * xv.y + xv.z * xv.z + xv.w * xv.w;
#pragma unroll
  for (int o = 16; o > 0; o >>= 1) {
    s  += __shfl_xor(s, o, 32);
    ss += __shfl_xor(ss, o, 32);
  }
  __shared__ float as[4], ass[4];
  if (lane_id() == 0) { as[t >> 5] = s; ass[t >> 5] = ss; }
  __syncthreads();
  s  = as[0] + as[1] + as[2] + as[3];
  ss = ass[0] + ass[1] + ass[2] + ass[3];
  float mu   = s / (float)D_;
  float var  = ss / (float)D_ - mu * mu;
  float rstd = rsqrtf(var + 1e-5f);
  int c = t * 4;
  const float4 wv = reinterpret_cast<const float4*>(w)[t];
  const float4 bv = reinterpret_cast<const float4*>(bia)[t];
  union { uint2 u; _Float16 h[4]; } pk;
  pk.h[0] = (_Float16)((xv.x - mu) * rstd * wv.x + bv.x);
  pk.h[1] = (_Float16)((xv.y - mu) * rstd * wv.y + bv.y);
  pk.h[2] = (_Float16)((xv.z - mu) * rstd * wv.z + bv.z);
  pk.h[3] = (_Float16)((xv.w - mu) * rstd * wv.w + bv.w);
  *reinterpret_cast<uint2*>(out + (size_t)row * D_ + c) = pk.u;
}

// Stage the 16-row A strip (16 x KD f16, contiguous) into LDS via async copy.
template <int KD>
__device__ __forceinline__ void stage_A(const _Float16* __restrict__ gA, _Float16* lA) {
  const int nchunks = (16 * KD) / 8;   // 16-byte chunks
  for (int i = threadIdx.x; i < nchunks; i += 256)
    async_copy16(gA + i * 8, lA + i * 8);
  async_wait0();
  __syncthreads();
}

// ---------------- K2: qvk = relu(xn @ qvk_w + b); split into Q,K (row-major f16)
// and V transposed per head: vT[b,h,d,s]. Block = 8 waves, tile 16x128.
__global__ __launch_bounds__(256) void gemm_qvk(const _Float16* __restrict__ xn,
                                                const _Float16* __restrict__ wT,
                                                const float* __restrict__ bias,
                                                _Float16* __restrict__ Qb,
                                                _Float16* __restrict__ Kb,
                                                _Float16* __restrict__ vT) {
  __shared__ _Float16 As[16 * D_];
  int w = threadIdx.x >> 5;
  int m0 = blockIdx.x * 16;
  int n0 = blockIdx.y * 128 + w * 16;
  stage_A<D_>(xn + (size_t)m0 * D_, As);
  FragC c; frag_zero(c);
  for (int k = 0; k < D_; k += 32) {
    v16h a = frag_ld(As + k, D_);
    v16h b = frag_ld(wT + (size_t)n0 * D_ + k, D_);
    c.v = WMMA_F16(a, b, c.v);
  }
  int lane = lane_id();
  int N = lane & 15;
  int mbase = m0 + ((lane & 16) ? 8 : 0);
  int cg = n0 + N;
  float bb = bias[cg];
  if (cg < 2 * D_) {
    _Float16* dst = (cg < D_) ? Qb : Kb;
    int cc = cg & (D_ - 1);
#pragma unroll
    for (int r = 0; r < 8; ++r)
      dst[(size_t)(mbase + r) * D_ + cc] = (_Float16)fmaxf(c.f[r] + bb, 0.0f);
  } else {
    int d = cg - 2 * D_;
    int h = d >> 6, dh = d & 63;
    int b  = mbase >> 11;       // row / S_
    int s0 = mbase & (S_ - 1);
    union { uint4 q; _Float16 hh[8]; } pk;
#pragma unroll
    for (int r = 0; r < 8; ++r) pk.hh[r] = (_Float16)fmaxf(c.f[r] + bb, 0.0f);
    *reinterpret_cast<uint4*>(vT + (((size_t)b * H_ + h) * DH_ + dh) * S_ + s0) = pk.q;
  }
}

// ---------------- K3: fused attention. One block per (b,h,16-row strip).
// qk (WMMA) -> LDS strip -> map_norm+sinmax (wave-local) -> scores out + hv (WMMA).
__global__ __launch_bounds__(256) void attn_fused(const _Float16* __restrict__ Qb,
                                                  const _Float16* __restrict__ Kb,
                                                  const _Float16* __restrict__ vT,
                                                  float* __restrict__ scores_out,
                                                  _Float16* __restrict__ hv) {
  extern __shared__ float strip[];            // 16 x 2048 f32 = 128 KB
  __shared__ float part[4][256];              // phase-5 split-K partials
  int bid  = blockIdx.x;
  int mblk = bid & 127;
  int h    = (bid >> 7) & 7;
  int b    = bid >> 10;
  int m0   = mblk * 16;
  int w    = threadIdx.x >> 5;
  int lane = lane_id();

  // Phase 1: qk strip. A = q rows (shared by all tiles of this strip).
  const _Float16* qbase = Qb + ((size_t)(b * S_ + m0)) * D_ + h * DH_;
  v16h a0 = frag_ld(qbase, D_);
  v16h a1 = frag_ld(qbase + 32, D_);
  for (int j = w; j < 128; j += 8) {
    int n0 = j * 16;
    const _Float16* kb = Kb + ((size_t)(b * S_ + n0)) * D_ + h * DH_;
    v16h b0 = frag_ld(kb, D_);
    v16h b1 = frag_ld(kb + 32, D_);
    FragC c; frag_zero(c);
    c.v = WMMA_F16(a0, b0, c.v);
    c.v = WMMA_F16(a1, b1, c.v);
    int N = lane & 15, Mb = (lane & 16) ? 8 : 0;
#pragma unroll
    for (int r = 0; r < 8; ++r) strip[(size_t)(Mb + r) * S_ + n0 + N] = c.f[r];
  }
  __syncthreads();

  // Phases 2-4: each wave owns 2 rows -> purely wave-local reductions.
#pragma unroll
  for (int rr = 0; rr < 2; ++rr) {
    int row = w * 2 + rr;
    float* rowp = strip + (size_t)row * S_;
    float s = 0.0f, ss = 0.0f;
    for (int ci = lane; ci < S_; ci += 32) {
      float xv = rowp[ci];
      s += xv; ss += xv * xv;
    }
#pragma unroll
    for (int o = 16; o > 0; o >>= 1) {
      s  += __shfl_xor(s, o, 32);
      ss += __shfl_xor(ss, o, 32);
    }
    float mu   = s / (float)S_;
    float var  = (ss - s * mu) / (float)(S_ - 1);   // ddof = 1
    float rstd = rsqrtf(var + 1e-5f);
    float ts = 0.0f;
    for (int ci = lane; ci < S_; ci += 32) {
      float z = (rowp[ci] - mu) * rstd;
      float t = 1.0f + __sinf(1.25f * z);           // 10/sqrt(DH) = 1.25
      rowp[ci] = t;
      ts += t;
    }
#pragma unroll
    for (int o = 16; o > 0; o >>= 1) ts += __shfl_xor(ts, o, 32);
    float inv = 1.0f / ts;
    size_t gbase = ((size_t)((b * H_ + h) * S_ + m0 + row)) * S_;
    for (int ci = lane; ci < S_; ci += 32) {
      float sc = rowp[ci] * inv;
      rowp[ci] = sc;
      scores_out[gbase + ci] = sc;                  // coalesced 128B/wave
    }
  }
  __syncthreads();

  // Phase 5: hv = scores(16x2048) @ v(2048x64). 8 waves: tile = w&3, k-half = w>>2.
  {
    int n0 = (w & 3) * 16;
    int k0 = (w >> 2) * (S_ / 2);
    FragC c; frag_zero(c);
    const _Float16* vb = vT + ((size_t)((b * H_ + h) * DH_ + n0)) * S_;
    for (int k = k0; k < k0 + S_ / 2; k += 32) {
      v16h af = frag_ld_lds_f32(strip, k);
      v16h bf = frag_ld(vb + k, S_);
      c.v = WMMA_F16(af, bf, c.v);
    }
    if (w >= 4) {
#pragma unroll
      for (int r = 0; r < 8; ++r) part[w - 4][r * 32 + lane] = c.f[r];
    }
    __syncthreads();
    if (w < 4) {
      int N = lane & 15, Mb = (lane & 16) ? 8 : 0;
#pragma unroll
      for (int r = 0; r < 8; ++r) {
        float v = c.f[r] + part[w][r * 32 + lane];
        hv[((size_t)(b * S_ + m0 + Mb + r)) * D_ + h * DH_ + n0 + N] = (_Float16)v;
      }
    }
  }
}

// ---------------- K4/K7: out_f32 = A_f16 @ wT + bias + res_f32  (Dout = 512)
template <int KD>
__global__ __launch_bounds__(256) void gemm_bias_res(const _Float16* __restrict__ A,
                                                     const _Float16* __restrict__ wT,
                                                     const float* __restrict__ bias,
                                                     const float* __restrict__ res,
                                                     float* __restrict__ out) {
  __shared__ _Float16 As[16 * KD];
  int w = threadIdx.x >> 5;
  int m0 = blockIdx.x * 16;
  int n0 = blockIdx.y * 128 + w * 16;
  stage_A<KD>(A + (size_t)m0 * KD, As);
  FragC c; frag_zero(c);
  for (int k = 0; k < KD; k += 32) {
    v16h a = frag_ld(As + k, KD);
    v16h b = frag_ld(wT + (size_t)n0 * KD + k, KD);
    c.v = WMMA_F16(a, b, c.v);
  }
  int lane = lane_id();
  int N = lane & 15;
  int mbase = m0 + ((lane & 16) ? 8 : 0);
  int cg = n0 + N;
  float bb = bias[cg];
#pragma unroll
  for (int r = 0; r < 8; ++r) {
    size_t idx = (size_t)(mbase + r) * D_ + cg;
    out[idx] = c.f[r] + bb + res[idx];
  }
}

// ---------------- K6: h = gelu_exact(a @ mlp1_w + b), f16 out [4096,1024]
__global__ __launch_bounds__(256) void gemm_gelu(const _Float16* __restrict__ A,
                                                 const _Float16* __restrict__ wT,
                                                 const float* __restrict__ bias,
                                                 _Float16* __restrict__ out) {
  __shared__ _Float16 As[16 * D_];
  int w = threadIdx.x >> 5;
  int m0 = blockIdx.x * 16;
  int n0 = blockIdx.y * 128 + w * 16;
  stage_A<D_>(A + (size_t)m0 * D_, As);
  FragC c; frag_zero(c);
  for (int k = 0; k < D_; k += 32) {
    v16h a = frag_ld(As + k, D_);
    v16h b = frag_ld(wT + (size_t)n0 * D_ + k, D_);
    c.v = WMMA_F16(a, b, c.v);
  }
  int lane = lane_id();
  int N = lane & 15;
  int mbase = m0 + ((lane & 16) ? 8 : 0);
  int cg = n0 + N;
  float bb = bias[cg];
#pragma unroll
  for (int r = 0; r < 8; ++r) {
    float v = c.f[r] + bb;
    float g = 0.5f * v * (1.0f + erff(v * 0.70710678118654752f));  // exact gelu
    out[(size_t)(mbase + r) * (2 * D_) + cg] = (_Float16)g;
  }
}

// ---------------- host side ----------------
extern "C" void kernel_launch(void* const* d_in, const int* in_sizes, int n_in,
                              void* d_out, int out_size, void* d_ws, size_t ws_size,
                              hipStream_t stream) {
  const float* x          = (const float*)d_in[0];
  const float* norm_w     = (const float*)d_in[1];
  const float* norm_b     = (const float*)d_in[2];
  const float* qvk_w      = (const float*)d_in[3];
  const float* qvk_b      = (const float*)d_in[4];
  const float* concat_w   = (const float*)d_in[5];
  const float* concat_b   = (const float*)d_in[6];
  const float* mlp_norm_w = (const float*)d_in[7];
  const float* mlp_norm_b = (const float*)d_in[8];
  const float* mlp1_w     = (const float*)d_in[9];
  const float* mlp1_b     = (const float*)d_in[10];
  const float* mlp2_w     = (const float*)d_in[11];
  const float* mlp2_b     = (const float*)d_in[12];

  float* mlpout = (float*)d_out;                                  // [B,S,D]
  float* scores = (float*)d_out + (size_t)ROWS_ * D_;             // [B,H,S,S]

  char* ws = (char*)d_ws;
  size_t off = 0;
  _Float16* wqvkT   = (_Float16*)(ws + off); off += (size_t)(3 * D_) * D_ * 2;      // [1536,512]
  _Float16* concatT = (_Float16*)(ws + off); off += (size_t)D_ * D_ * 2;            // [512,512]
  _Float16* mlp1T   = (_Float16*)(ws + off); off += (size_t)(2 * D_) * D_ * 2;      // [1024,512]
  _Float16* mlp2T   = (_Float16*)(ws + off); off += (size_t)D_ * (2 * D_) * 2;      // [512,1024]
  _Float16* xn      = (_Float16*)(ws + off); off += (size_t)ROWS_ * D_ * 2;
  _Float16* Qb      = (_Float16*)(ws + off); off += (size_t)ROWS_ * D_ * 2;
  _Float16* Kb      = (_Float16*)(ws + off); off += (size_t)ROWS_ * D_ * 2;
  _Float16* vT      = (_Float16*)(ws + off); off += (size_t)B_ * H_ * DH_ * S_ * 2;
  _Float16* hv      = (_Float16*)(ws + off); off += (size_t)ROWS_ * D_ * 2;
  float*    attb    = (float*)(ws + off);    off += (size_t)ROWS_ * D_ * 4;
  _Float16* abuf    = (_Float16*)(ws + off); off += (size_t)ROWS_ * D_ * 2;
  _Float16* hbuf    = (_Float16*)(ws + off); off += (size_t)ROWS_ * (2 * D_) * 2;
  (void)ws_size; (void)n_in; (void)in_sizes; (void)out_size;

  // weight transposition + f16 conversion
  wconv<<<(D_ * 3 * D_ + 255) / 256, 256, 0, stream>>>(qvk_w, wqvkT, D_, 3 * D_);
  wconv<<<(D_ * D_ + 255) / 256, 256, 0, stream>>>(concat_w, concatT, D_, D_);
  wconv<<<(D_ * 2 * D_ + 255) / 256, 256, 0, stream>>>(mlp1_w, mlp1T, D_, 2 * D_);
  wconv<<<(2 * D_ * D_ + 255) / 256, 256, 0, stream>>>(mlp2_w, mlp2T, 2 * D_, D_);

  // xn = layernorm(x)
  ln512<<<ROWS_, 128, 0, stream>>>(x, norm_w, norm_b, xn);

  // Q,K,V = relu(xn @ qvk_w + b)
  gemm_qvk<<<dim3(ROWS_ / 16, (3 * D_) / 128), 256, 0, stream>>>(xn, wqvkT, qvk_b, Qb, Kb, vT);

  // fused attention: qk -> map_norm -> sinmax -> scores + hv
  attn_fused<<<B_ * H_ * (S_ / 16), 256, 16 * S_ * sizeof(float), stream>>>(Qb, Kb, vT, scores, hv);

  // attout = hv @ concat_w + b + x (residual)
  gemm_bias_res<D_><<<dim3(ROWS_ / 16, D_ / 128), 256, 0, stream>>>(hv, concatT, concat_b, x, attb);

  // a = layernorm(attout)
  ln512<<<ROWS_, 128, 0, stream>>>(attb, mlp_norm_w, mlp_norm_b, abuf);

  // h = gelu(a @ mlp1_w + b)
  gemm_gelu<<<dim3(ROWS_ / 16, (2 * D_) / 128), 256, 0, stream>>>(abuf, mlp1T, mlp1_b, hbuf);

  // mlpout = h @ mlp2_w + b + attout
  gemm_bias_res<2 * D_><<<dim3(ROWS_ / 16, D_ / 128), 256, 0, stream>>>(hbuf, mlp2T, mlp2_b, attb, mlpout);
}
